// MultiheadAttention_23562190586279
// MI455X (gfx1250) — matrix operand order repaired and checked by reading, not compiled, not run on hardware
//
#include <hip/hip_runtime.h>

typedef __attribute__((ext_vector_type(16))) __bf16 v16bf;
typedef __attribute__((ext_vector_type(8)))  __bf16 v8bf;
typedef __attribute__((ext_vector_type(8)))  float  v8f;
typedef __attribute__((ext_vector_type(4)))  unsigned u32x4;
typedef __attribute__((ext_vector_type(8)))  int      i32x8;
typedef __attribute__((ext_vector_type(4)))  int      i32x4;

#define N_HEAD 16
#define LSEQ   2048
#define DMODEL 1024
#define DHEAD  64
#define BATCH  2
#define MROWS  (BATCH * LSEQ)   /* 4096 */

#if defined(__has_builtin)
#if __has_builtin(__builtin_amdgcn_tensor_load_to_lds) && \
    __has_builtin(__builtin_amdgcn_s_wait_tensorcnt)
#define HAVE_TDM 1
#endif
#endif

#define WMMA_BF16(a, b, c) \
  __builtin_amdgcn_wmma_f32_16x16x32_bf16(false, (a), false, (b), (short)0, (c), false, false)

static __device__ __forceinline__ __bf16 f2bf(float f) {
  unsigned u = __builtin_bit_cast(unsigned, f);
  unsigned r = (u + 0x7FFFu + ((u >> 16) & 1u)) >> 16;  // round-to-nearest-even
  unsigned short s = (unsigned short)r;
  return __builtin_bit_cast(__bf16, s);
}

// -------------------- bulk f32 -> bf16 conversion (bandwidth bound) --------------------
__global__ __launch_bounds__(256) void cvt_f32_bf16_kernel(const float* __restrict__ src,
                                                           __bf16* __restrict__ dst, int n8) {
  int i = blockIdx.x * blockDim.x + threadIdx.x;
  if (i >= n8) return;
  const float* p = src + (size_t)i * 8;
  float4 a = *(const float4*)p;
  float4 b = *(const float4*)(p + 4);
  v8bf r;
  r[0] = f2bf(a.x); r[1] = f2bf(a.y); r[2] = f2bf(a.z); r[3] = f2bf(a.w);
  r[4] = f2bf(b.x); r[5] = f2bf(b.y); r[6] = f2bf(b.z); r[7] = f2bf(b.w);
  *(v8bf*)(dst + (size_t)i * 8) = r;
}

// A-operand fragment (16x32 bf16): lane half hh gets 16 contiguous K values [16h,16h+16)
static __device__ __forceinline__ v16bf load_afrag(const __bf16* row, int kk, int hh) {
  union { v16bf v; v8bf c[2]; } u;
  u.c[0] = *(const v8bf*)(row + kk + 16 * hh);
  u.c[1] = *(const v8bf*)(row + kk + 8 + 16 * hh);
  return u.v;
}
// B-operand fragment (32x16 bf16): lane (col n) reads 16 contiguous K values at 16*hh
static __device__ __forceinline__ v16bf load_bfrag(const __bf16* p) {
  return *(const v16bf*)p;
}

#ifdef HAVE_TDM
// Issue a 2D TDM tile load: tile_d1 rows x tile_d0 elements (bf16), row stride stride0,
// compacted into LDS at lds_off. D# packing per CDNA5 ISA 8.3/8.4.
static __device__ __forceinline__ void tdm_load_2d(unsigned lds_off, const void* gptr,
                                                   unsigned tensor_d0, unsigned tensor_d1,
                                                   unsigned tile_d0, unsigned tile_d1,
                                                   unsigned stride0) {
  unsigned long long ga = (unsigned long long)(uintptr_t)gptr;
  u32x4 g0;
  g0[0] = 1u;                                             // count=1, user descriptor
  g0[1] = lds_off;                                        // LDS byte address
  g0[2] = (unsigned)ga;                                   // global_addr[31:0]
  g0[3] = (unsigned)((ga >> 32) & 0x01FFFFFFu) | (2u << 30);  // addr[56:32] | type=2
  i32x8 g1;
  g1[0] = (int)(1u << 16);                                // wg_mask=0, data_size=2B
  g1[1] = (int)((tensor_d0 & 0xFFFFu) << 16);             // tensor_dim0[15:0] @ [63:48]
  g1[2] = (int)((tensor_d0 >> 16) | ((tensor_d1 & 0xFFFFu) << 16));
  g1[3] = (int)((tensor_d1 >> 16) | (tile_d0 << 16));     // tile_dim0 @ [127:112]
  g1[4] = (int)tile_d1;                                   // tile_dim1 @ [143:128], tile_dim2=0
  g1[5] = (int)stride0;                                   // tensor_dim0_stride[31:0]
  g1[6] = 0;                                              // stride0[47:32], stride1[15:0]
  g1[7] = 0;
  i32x4 z4 = {0, 0, 0, 0};
  i32x8 z8 = {0, 0, 0, 0, 0, 0, 0, 0};
  __builtin_amdgcn_tensor_load_to_lds(g0, g1, z4, z4, z8, 0);
}
#endif

// -------------------- projection: Y = X @ W^T (all bf16 in, f32 acc) --------------------
// Wave tile: 32 (M) x 64 (N).  Block: 4 waves stacked along M -> 128 x 64.
// MODE 0: out bf16 row-major [MROWS][DMODEL]
// MODE 1: out bf16 per-head transposed V: [b][h][d][L]
// MODE 2: out f32 row-major [MROWS][DMODEL]
template <int MODE>
__global__ __launch_bounds__(128) void proj_kernel(const __bf16* __restrict__ X,
                                                   const __bf16* __restrict__ W,
                                                   void* __restrict__ outv) {
  const int wave = threadIdx.x >> 5, lane = threadIdx.x & 31;
  const int hh = lane >> 4, ln = lane & 15;
  const int n0 = blockIdx.x * 64;
  const int m0 = blockIdx.y * 128 + wave * 32;

  const __bf16* arow0 = X + (size_t)(m0 + ln) * DMODEL;
  const __bf16* arow1 = X + (size_t)(m0 + 16 + ln) * DMODEL;
  v8f acc[2][4] = {};

  for (int kk = 0; kk < DMODEL; kk += 32) {
    v16bf a0 = load_afrag(arow0, kk, hh);
    v16bf a1 = load_afrag(arow1, kk, hh);
    v16bf b[4];
#pragma unroll
    for (int t = 0; t < 4; t++)
      b[t] = load_bfrag(W + (size_t)(n0 + t * 16 + ln) * DMODEL + kk + 16 * hh);
#pragma unroll
    for (int t = 0; t < 4; t++) {
      acc[0][t] = WMMA_BF16(a0, b[t], acc[0][t]);
      acc[1][t] = WMMA_BF16(a1, b[t], acc[1][t]);
    }
  }

#pragma unroll
  for (int g = 0; g < 2; g++) {
#pragma unroll
    for (int t = 0; t < 4; t++) {
#pragma unroll
      for (int j = 0; j < 8; j++) {
        const int m = m0 + 16 * g + j + 8 * hh;  // C layout: row = VGPRj + 8*(lane>>4)
        const int n = n0 + t * 16 + ln;          // col = lane&15
        const float val = acc[g][t][j];
        if (MODE == 0) {
          ((__bf16*)outv)[(size_t)m * DMODEL + n] = f2bf(val);
        } else if (MODE == 1) {
          const int bb = m >> 11, l = m & (LSEQ - 1);
          const int hd = n >> 6, d = n & (DHEAD - 1);
          ((__bf16*)outv)[(((size_t)bb * N_HEAD + hd) * DHEAD + d) * LSEQ + l] = f2bf(val);
        } else {
          ((float*)outv)[(size_t)m * DMODEL + n] = val;
        }
      }
    }
  }
}

// -------------------- flash attention per (b, h, 64-row q-tile) --------------------
// K/V tiles (64x64 bf16 each) are DMA'd into LDS by the Tensor Data Mover,
// double-buffered, and shared by all 4 waves of the block.
__global__ __launch_bounds__(128) void attn_fwd_kernel(const __bf16* __restrict__ qp,
                                                       const __bf16* __restrict__ kp,
                                                       const __bf16* __restrict__ vpT,
                                                       __bf16* __restrict__ attn) {
  const int qt = blockIdx.x, h = blockIdx.y, b = blockIdx.z;
  const int wave = threadIdx.x >> 5, lane = threadIdx.x & 31;
  const int hh = lane >> 4, ln = lane & 15;
  const int m0 = qt * 64 + wave * 16;
  const int NB = LSEQ / 64;

  const __bf16* qbase = qp + (size_t)b * LSEQ * DMODEL + (size_t)h * DHEAD;
  const __bf16* kbase = kp + (size_t)b * LSEQ * DMODEL + (size_t)h * DHEAD;
  const __bf16* vbase = vpT + ((size_t)b * N_HEAD + h) * DHEAD * LSEQ;

  __shared__ __bf16 kvlds[2][2][64 * 64];  // [buf][K/V][tile] : 32 KB
  __shared__ __bf16 plds[4][16 * 64];      // per-wave P staging : 8 KB
  __bf16* myp = &plds[wave][0];

  // Q fragments (row of 64 bf16 -> two K=32 A-frags), loaded once
  v16bf aq[2];
#pragma unroll
  for (int s = 0; s < 2; s++)
    aq[s] = load_afrag(qbase + (size_t)(m0 + ln) * DMODEL, 32 * s, hh);

  v8f o[4] = {};
  float mrow[8], lrow[8];
#pragma unroll
  for (int j = 0; j < 8; j++) { mrow[j] = -3.0e38f; lrow[j] = 0.0f; }

#ifdef HAVE_TDM
  if (wave == 0) {
    // prime buffer 0: K tile -> LDS[kj][d], V tile -> LDS[d][kj]
    tdm_load_2d((unsigned)(uintptr_t)&kvlds[0][0][0], kbase,
                DMODEL, LSEQ, DHEAD, 64, DMODEL);
    tdm_load_2d((unsigned)(uintptr_t)&kvlds[0][1][0], vbase,
                LSEQ, DHEAD, 64, DHEAD, LSEQ);
  }
#endif

  for (int kb = 0; kb < NB; ++kb) {
    const int buf = kb & 1;
#ifdef HAVE_TDM
    if (kb) __syncthreads();  // all waves done reading buf^1 before TDM overwrites it
    if (kb + 1 < NB) {
      if (wave == 0) {
        const int kc2 = (kb + 1) * 64;
        tdm_load_2d((unsigned)(uintptr_t)&kvlds[buf ^ 1][0][0],
                    kbase + (size_t)kc2 * DMODEL, DMODEL, LSEQ, DHEAD, 64, DMODEL);
        tdm_load_2d((unsigned)(uintptr_t)&kvlds[buf ^ 1][1][0],
                    vbase + kc2, LSEQ, DHEAD, 64, DHEAD, LSEQ);
        __builtin_amdgcn_s_wait_tensorcnt(2);  // current tile done, next 2 in flight
      }
    } else {
      if (wave == 0) __builtin_amdgcn_s_wait_tensorcnt(0);
    }
    __syncthreads();
#else
    // fallback: cooperative load of K/V tiles into LDS buffer 0
    const int kc = kb * 64;
    const int r = threadIdx.x >> 1, c0 = (threadIdx.x & 1) * 32;
    __syncthreads();
    *(v16bf*)&kvlds[0][0][r * 64 + c0] =
        *(const v16bf*)(kbase + (size_t)(kc + r) * DMODEL + c0);
    *(v16bf*)&kvlds[0][0][r * 64 + c0 + 16] =
        *(const v16bf*)(kbase + (size_t)(kc + r) * DMODEL + c0 + 16);
    *(v16bf*)&kvlds[0][1][r * 64 + c0] =
        *(const v16bf*)(vbase + (size_t)r * LSEQ + kc + c0);
    *(v16bf*)&kvlds[0][1][r * 64 + c0 + 16] =
        *(const v16bf*)(vbase + (size_t)r * LSEQ + kc + c0 + 16);
    __syncthreads();
#endif
    const __bf16* kt = &kvlds[
#ifdef HAVE_TDM
        buf
#else
        0
#endif
    ][0][0];
    const __bf16* vt = kt + 64 * 64;

    // S = Q @ K^T for this wave's 16 rows x 64 keys
    v16bf bk[4][2];
#pragma unroll
    for (int t = 0; t < 4; t++)
#pragma unroll
      for (int s = 0; s < 2; s++)
        bk[t][s] = load_bfrag(kt + (t * 16 + ln) * 64 + 32 * s + 16 * hh);

    v8f sacc[4];
#pragma unroll
    for (int t = 0; t < 4; t++) {
      v8f c = {};
      c = WMMA_BF16(aq[0], bk[t][0], c);
      c = WMMA_BF16(aq[1], bk[t][1], c);
      sacc[t] = c;
    }

    // V fragments now so the DS loads overlap the softmax VALU/exp work
    v16bf bv[4][2];
#pragma unroll
    for (int t = 0; t < 4; t++)
#pragma unroll
      for (int s = 0; s < 2; s++)
        bv[t][s] = load_bfrag(vt + (t * 16 + ln) * 64 + 32 * s + 16 * hh);

    // online softmax over this 64-column block
#pragma unroll
    for (int j = 0; j < 8; j++) {
      float s0 = sacc[0][j] * 0.125f;   // 1/sqrt(64)
      float s1 = sacc[1][j] * 0.125f;
      float s2 = sacc[2][j] * 0.125f;
      float s3 = sacc[3][j] * 0.125f;
      float bm = fmaxf(fmaxf(s0, s1), fmaxf(s2, s3));
#pragma unroll
      for (int msk = 1; msk < 16; msk <<= 1)
        bm = fmaxf(bm, __shfl_xor(bm, msk, 32));   // stays within 16-lane half
      float mnew = fmaxf(mrow[j], bm);
      float alpha = __expf(mrow[j] - mnew);
      mrow[j] = mnew;
      float p0 = __expf(s0 - mnew);
      float p1 = __expf(s1 - mnew);
      float p2 = __expf(s2 - mnew);
      float p3 = __expf(s3 - mnew);
      float rs = (p0 + p1) + (p2 + p3);
#pragma unroll
      for (int msk = 1; msk < 16; msk <<= 1)
        rs += __shfl_xor(rs, msk, 32);
      lrow[j] = lrow[j] * alpha + rs;
#pragma unroll
      for (int t = 0; t < 4; t++) o[t][j] *= alpha;
      // stage P (C layout) into LDS row-major [16][64]
      const int rbase = (j + 8 * hh) * 64 + ln;
      myp[rbase + 0]  = f2bf(p0);
      myp[rbase + 16] = f2bf(p1);
      myp[rbase + 32] = f2bf(p2);
      myp[rbase + 48] = f2bf(p3);
    }

    // same-wave LDS ops are in-order; explicit wait for safety
    asm volatile("s_wait_dscnt 0" ::: "memory");

    // reload P as A-operand fragments
    v16bf ap[2];
#pragma unroll
    for (int s = 0; s < 2; s++) {
      union { v16bf v; v8bf c[2]; } u;
      u.c[0] = *(const v8bf*)(myp + ln * 64 + 32 * s + 16 * hh);
      u.c[1] = *(const v8bf*)(myp + ln * 64 + 32 * s + 8 + 16 * hh);
      ap[s] = u.v;
    }

    // O += P @ V
#pragma unroll
    for (int t = 0; t < 4; t++) {
      o[t] = WMMA_BF16(ap[0], bv[t][0], o[t]);
      o[t] = WMMA_BF16(ap[1], bv[t][1], o[t]);
    }
  }

  // normalize and write attn (bf16, row-major [b*L][DMODEL])
#pragma unroll
  for (int j = 0; j < 8; j++) {
    float inv = 1.0f / lrow[j];
    size_t row = (size_t)b * LSEQ + m0 + j + 8 * hh;
#pragma unroll
    for (int t = 0; t < 4; t++) {
      attn[row * DMODEL + h * DHEAD + t * 16 + ln] = f2bf(o[t][j] * inv);
    }
  }
}

extern "C" void kernel_launch(void* const* d_in, const int* in_sizes, int n_in,
                              void* d_out, int out_size, void* d_ws, size_t ws_size,
                              hipStream_t stream) {
  (void)in_sizes; (void)n_in; (void)out_size; (void)ws_size;
  const float* q  = (const float*)d_in[0];
  const float* k  = (const float*)d_in[1];
  const float* v  = (const float*)d_in[2];
  const float* Wq = (const float*)d_in[3];
  const float* Wk = (const float*)d_in[4];
  const float* Wv = (const float*)d_in[5];
  const float* Wo = (const float*)d_in[6];

  char* ws = (char*)d_ws;
  const size_t MB = 1024 * 1024;
  __bf16* qb   = (__bf16*)(ws + 0 * MB);    // [4096][1024] bf16 : 8 MB
  __bf16* kb   = (__bf16*)(ws + 8 * MB);
  __bf16* vb   = (__bf16*)(ws + 16 * MB);
  __bf16* Wqb  = (__bf16*)(ws + 24 * MB);   // [1024][1024] bf16 : 2 MB each
  __bf16* Wkb  = (__bf16*)(ws + 26 * MB);
  __bf16* Wvb  = (__bf16*)(ws + 28 * MB);
  __bf16* Wob  = (__bf16*)(ws + 30 * MB);
  __bf16* qp   = (__bf16*)(ws + 32 * MB);   // projected Q bf16
  __bf16* kp   = (__bf16*)(ws + 40 * MB);   // projected K bf16
  __bf16* vpT  = (__bf16*)(ws + 48 * MB);   // projected V, [b][h][d][L]
  __bf16* attn = (__bf16*)(ws + 56 * MB);   // attention output bf16

  // one-shot conversions (bandwidth bound, ~100 MB total traffic)
  const int nin8 = MROWS * DMODEL / 8;      // 524288
  const int nw8  = DMODEL * DMODEL / 8;     // 131072
  cvt_f32_bf16_kernel<<<(nin8 + 255) / 256, 256, 0, stream>>>(q,  qb,  nin8);
  cvt_f32_bf16_kernel<<<(nin8 + 255) / 256, 256, 0, stream>>>(k,  kb,  nin8);
  cvt_f32_bf16_kernel<<<(nin8 + 255) / 256, 256, 0, stream>>>(v,  vb,  nin8);
  cvt_f32_bf16_kernel<<<(nw8  + 255) / 256, 256, 0, stream>>>(Wq, Wqb, nw8);
  cvt_f32_bf16_kernel<<<(nw8  + 255) / 256, 256, 0, stream>>>(Wk, Wkb, nw8);
  cvt_f32_bf16_kernel<<<(nw8  + 255) / 256, 256, 0, stream>>>(Wv, Wvb, nw8);
  cvt_f32_bf16_kernel<<<(nw8  + 255) / 256, 256, 0, stream>>>(Wo, Wob, nw8);

  dim3 pg(DMODEL / 64, MROWS / 128), pb(128);
  proj_kernel<0><<<pg, pb, 0, stream>>>(qb, Wqb, (void*)qp);
  proj_kernel<0><<<pg, pb, 0, stream>>>(kb, Wkb, (void*)kp);
  proj_kernel<1><<<pg, pb, 0, stream>>>(vb, Wvb, (void*)vpT);
  attn_fwd_kernel<<<dim3(LSEQ / 64, N_HEAD, BATCH), pb, 0, stream>>>(qp, kp, vpT, attn);
  proj_kernel<2><<<pg, pb, 0, stream>>>(attn, Wob, d_out);
}
